// Transformer_9345848836434
// MI455X (gfx1250) — compile-verified, matrix operand order checked by reading, not run on hardware
//
#include <hip/hip_runtime.h>
#include <hip/hip_bf16.h>

// ---------------------------------------------------------------------------
// Model constants (match the reference)
// ---------------------------------------------------------------------------
#define SEQ      256
#define BATCH    8
#define N_NODES  (SEQ * BATCH)        // 2048
#define VOCAB    259
#define D_MODEL  256
#define NHEAD    8
#define D_K      32
#define D_FF     1024
#define N_LAYERS 2

typedef __attribute__((ext_vector_type(16))) _Float16 v16h;
typedef __attribute__((ext_vector_type(8)))  _Float16 v8h;
typedef __attribute__((ext_vector_type(8)))  float    v8f;

// ---------------------------------------------------------------------------
// f32 [K,N] -> f16 [N,K] (transpose + convert) so WMMA B fragments are
// contiguous 16B loads along K, just like A fragments.
// ---------------------------------------------------------------------------
__global__ void cvt_transpose_kernel(const float* __restrict__ in,
                                     _Float16* __restrict__ out,
                                     int K, int N) {
  int idx = blockIdx.x * blockDim.x + threadIdx.x;
  if (idx >= K * N) return;
  int n = idx / K;
  int k = idx - n * K;
  out[(size_t)n * K + k] = (_Float16)in[(size_t)k * N + n];
}

// ---------------------------------------------------------------------------
// x[n,d] = coord_emb[pos%3,d] + pos_emb[pos/3,d] + value_emb[tok,d]
// ---------------------------------------------------------------------------
__global__ void embed_kernel(const int* __restrict__ tokens,
                             const int* __restrict__ positions,
                             const float* __restrict__ coord,
                             const float* __restrict__ pose,
                             const float* __restrict__ val,
                             float* __restrict__ x) {
  int idx = blockIdx.x * blockDim.x + threadIdx.x;   // N_NODES * D_MODEL
  int node = idx >> 8;
  int d = idx & 255;
  int p = positions[node];
  int t = tokens[node];
  x[idx] = coord[(p % 3) * D_MODEL + d] +
           pose[(p / 3) * D_MODEL + d] +
           val[(size_t)t * D_MODEL + d];
}

// ---------------------------------------------------------------------------
// LayerNorm over 256 dims; one wave32 per row; emits f16 (WMMA A operand).
// ---------------------------------------------------------------------------
__global__ void ln_kernel(const float* __restrict__ x,
                          const float* __restrict__ scale,
                          const float* __restrict__ bias,
                          _Float16* __restrict__ out) {
  int row  = blockIdx.x * (blockDim.x >> 5) + (threadIdx.x >> 5);
  int lane = threadIdx.x & 31;
  const float* xp = x + (size_t)row * D_MODEL;
  float v[8];
  float sum = 0.f, sq = 0.f;
#pragma unroll
  for (int j = 0; j < 8; ++j) {
    v[j] = xp[lane + 32 * j];
    sum += v[j];
    sq  += v[j] * v[j];
  }
#pragma unroll
  for (int off = 16; off > 0; off >>= 1) {
    sum += __shfl_xor(sum, off, 32);
    sq  += __shfl_xor(sq,  off, 32);
  }
  float mu  = sum * (1.0f / D_MODEL);
  float var = sq * (1.0f / D_MODEL) - mu * mu;
  float r   = rsqrtf(var + 1e-5f);
  _Float16* op = out + (size_t)row * D_MODEL;
#pragma unroll
  for (int j = 0; j < 8; ++j) {
    int d = lane + 32 * j;
    op[d] = (_Float16)((v[j] - mu) * r * scale[d] + bias[d]);
  }
}

// ---------------------------------------------------------------------------
// WMMA GEMM: C[M,N] = act(A[M,K] @ Bt[N,K]^T + bias) (+ resid)
// One wave32 per 16(M) x 64(N) tile: 4 independent f32 accumulators share one
// A fragment -> 4x v_wmma_f32_16x16x32_f16 per K-step with no RAW hazards.
// K is a template constant (256 / 1024): the pipelined K-loop fully unrolls,
// so next-step fragment loads overlap the current wmma group (distinct regs,
// no forced s_wait_loadcnt 0 between wmmas).
// OOB columns (N=259 head GEMM) handled by CLAMPING the Bt row index: a WMMA
// output column depends only on its own B column, and OOB columns are never
// stored, so no zero-fill / exec branching is needed.
// A fragment layout (ISA 7.12.2, 16-bit A 16x32):
//   lanes 0-15 : row = lane,    K = {k0..k0+7, k0+16..k0+23}
//   lanes16-31 : row = lane-16, K = {k0+8..k0+15, k0+24..k0+31}
// B fragment mirrors A over columns of Bt.
// ---------------------------------------------------------------------------
template <int K>
__global__ void wmma_gemm_kernel(const _Float16* __restrict__ A,
                                 const _Float16* __restrict__ Bt,
                                 const float* __restrict__ bias,
                                 const float* __restrict__ resid,
                                 float* __restrict__ Cout,
                                 _Float16* __restrict__ Chalf,
                                 int N, int relu) {
  int lane  = threadIdx.x & 31;
  int tileN = blockIdx.x * 64;
  int tileM = blockIdx.y * 16;
  int arow  = tileM + (lane & 15);
  int khalf = (lane >> 4) << 3;           // 0 or 8

  const _Float16* ap = A + (size_t)arow * K + khalf;
  const _Float16* bp[4];
  int ncol[4];
#pragma unroll
  for (int j = 0; j < 4; ++j) {
    int c   = tileN + j * 16 + (lane & 15);
    ncol[j] = c;
    int cc  = (c < N) ? c : (N - 1);      // clamp; OOB columns never stored
    bp[j]   = Bt + (size_t)cc * K + khalf;
  }

  v8f acc[4];
#pragma unroll
  for (int j = 0; j < 4; ++j) acc[j] = (v8f){};

  auto loadA = [&](int k0) -> v16h {
    union { v16h v; v8h h[2]; } t;
    t.h[0] = *(const v8h*)(ap + k0);
    t.h[1] = *(const v8h*)(ap + k0 + 16);
    return t.v;
  };
  auto loadB = [&](int j, int k0) -> v16h {
    union { v16h v; v8h h[2]; } t;
    t.h[0] = *(const v8h*)(bp[j] + k0);
    t.h[1] = *(const v8h*)(bp[j] + k0 + 16);
    return t.v;
  };

  // software-pipelined, fully unrolled K loop
  v16h aCur = loadA(0);
  v16h bCur[4];
#pragma unroll
  for (int j = 0; j < 4; ++j) bCur[j] = loadB(j, 0);

#pragma unroll
  for (int k0 = 0; k0 < K - 32; k0 += 32) {
    v16h aNxt = loadA(k0 + 32);
    v16h bNxt[4];
#pragma unroll
    for (int j = 0; j < 4; ++j) bNxt[j] = loadB(j, k0 + 32);
#pragma unroll
    for (int j = 0; j < 4; ++j)
      acc[j] = __builtin_amdgcn_wmma_f32_16x16x32_f16(
          false, aCur, false, bCur[j], (short)0, acc[j], false, false);
    aCur = aNxt;
#pragma unroll
    for (int j = 0; j < 4; ++j) bCur[j] = bNxt[j];
  }
#pragma unroll
  for (int j = 0; j < 4; ++j)
    acc[j] = __builtin_amdgcn_wmma_f32_16x16x32_f16(
        false, aCur, false, bCur[j], (short)0, acc[j], false, false);

  // D layout: VGPR r -> M = r (lanes 0-15) or 8+r (lanes 16-31), N = lane&15
  int m0 = tileM + ((lane >> 4) << 3);
#pragma unroll
  for (int j = 0; j < 4; ++j) {
    int n = ncol[j];
    if (n < N) {
      float bval = bias ? bias[n] : 0.0f;
#pragma unroll
      for (int r = 0; r < 8; ++r) {
        float vo = acc[j][r] + bval;
        if (relu) vo = fmaxf(vo, 0.0f);
        size_t off = (size_t)(m0 + r) * N + n;
        if (resid) vo += resid[off];
        if (Cout)  Cout[off]  = vo;
        if (Chalf) Chalf[off] = (_Float16)vo;
      }
    }
  }
}

// ---------------------------------------------------------------------------
// Causal attention. qkv layout per node: [3][H][D_K] at stride 768.
// One block (256 thr) per dst node; wave h handles head h.
// Two-pass softmax with scores staged in LDS; alpha*V reduced by shfl butterfly.
// Emits f16 z (WMMA A operand for the Wo GEMM).
// ---------------------------------------------------------------------------
__global__ void attn_kernel(const float* __restrict__ qkv,
                            _Float16* __restrict__ zout) {
  __shared__ float s_scores[NHEAD][SEQ];
  __shared__ float s_q[NHEAD][D_K];
  int node = blockIdx.x;
  int h    = threadIdx.x >> 5;
  int lane = threadIdx.x & 31;
  int pos  = node & (SEQ - 1);
  int base = node - pos;
  int nsrc = pos + 1;
  const float scale = 0.17677669529663687f;  // 1/sqrt(D_K)

  s_q[h][lane] = qkv[(size_t)node * 768 + h * D_K + lane];
  __syncthreads();

  // pass 1: scores + running max
  float lmax = -1e30f;
  for (int s = lane; s < nsrc; s += 32) {
    const float* kp = qkv + (size_t)(base + s) * 768 + 256 + h * D_K;
    float a = 0.f;
#pragma unroll
    for (int d = 0; d < D_K; ++d) a += s_q[h][d] * kp[d];
    a *= scale;
    s_scores[h][s] = a;
    lmax = fmaxf(lmax, a);
  }
#pragma unroll
  for (int off = 16; off > 0; off >>= 1)
    lmax = fmaxf(lmax, __shfl_xor(lmax, off, 32));

  // denom
  float lsum = 0.f;
  for (int s = lane; s < nsrc; s += 32)
    lsum += __expf(s_scores[h][s] - lmax);
#pragma unroll
  for (int off = 16; off > 0; off >>= 1)
    lsum += __shfl_xor(lsum, off, 32);
  float inv = 1.0f / lsum;

  // pass 2: z = sum alpha * v
  float zv[D_K];
#pragma unroll
  for (int d = 0; d < D_K; ++d) zv[d] = 0.f;
  for (int s = lane; s < nsrc; s += 32) {
    float alpha = __expf(s_scores[h][s] - lmax) * inv;
    const float* vp = qkv + (size_t)(base + s) * 768 + 512 + h * D_K;
#pragma unroll
    for (int d = 0; d < D_K; ++d) zv[d] += alpha * vp[d];
  }
#pragma unroll
  for (int d = 0; d < D_K; ++d) {
#pragma unroll
    for (int off = 16; off > 0; off >>= 1)
      zv[d] += __shfl_xor(zv[d], off, 32);
  }
  if (lane == 0) {
    _Float16* zp = zout + (size_t)node * D_MODEL + h * D_K;
#pragma unroll
    for (int d = 0; d < D_K; ++d) zp[d] = (_Float16)zv[d];
  }
}

// ---------------------------------------------------------------------------
// In-place log_softmax over VOCAB=259 per row; one wave per row.
// ---------------------------------------------------------------------------
__global__ void logsoftmax_kernel(float* __restrict__ out, int N) {
  int row  = blockIdx.x * (blockDim.x >> 5) + (threadIdx.x >> 5);
  int lane = threadIdx.x & 31;
  float* p = out + (size_t)row * N;
  float vals[9];
  float m = -1e30f;
  int cnt = 0;
  for (int i = lane; i < N; i += 32) {
    vals[cnt] = p[i];
    m = fmaxf(m, vals[cnt]);
    ++cnt;
  }
#pragma unroll
  for (int off = 16; off > 0; off >>= 1)
    m = fmaxf(m, __shfl_xor(m, off, 32));
  float s = 0.f;
  for (int c = 0; c < cnt; ++c) s += __expf(vals[c] - m);
#pragma unroll
  for (int off = 16; off > 0; off >>= 1)
    s += __shfl_xor(s, off, 32);
  float lse = m + __logf(s);
  cnt = 0;
  for (int i = lane; i < N; i += 32) {
    p[i] = vals[cnt] - lse;
    ++cnt;
  }
}

// ---------------------------------------------------------------------------
// Host orchestration
// ---------------------------------------------------------------------------
static inline void launch_gemm(const _Float16* A, const _Float16* Bt,
                               const float* bias, const float* resid,
                               float* Cout, _Float16* Chalf,
                               int M, int N, int K, int relu,
                               hipStream_t stream) {
  dim3 grid((N + 63) / 64, M / 16);
  if (K == D_MODEL) {
    wmma_gemm_kernel<D_MODEL><<<grid, 32, 0, stream>>>(A, Bt, bias, resid,
                                                       Cout, Chalf, N, relu);
  } else {
    wmma_gemm_kernel<D_FF><<<grid, 32, 0, stream>>>(A, Bt, bias, resid,
                                                    Cout, Chalf, N, relu);
  }
}

extern "C" void kernel_launch(void* const* d_in, const int* in_sizes, int n_in,
                              void* d_out, int out_size, void* d_ws, size_t ws_size,
                              hipStream_t stream) {
  (void)in_sizes; (void)n_in; (void)out_size; (void)ws_size;

  const int*   tokens    = (const int*)d_in[0];
  const int*   positions = (const int*)d_in[1];
  // d_in[2] edge_src, d_in[3] edge_dst: structurally the causal tril mask -> unused
  const float* coord_emb = (const float*)d_in[4];
  const float* pos_emb   = (const float*)d_in[5];
  const float* value_emb = (const float*)d_in[6];
  const float* ln1_scale = (const float*)d_in[7];
  const float* ln1_bias  = (const float*)d_in[8];
  const float* Wqkv      = (const float*)d_in[9];
  const float* Wo        = (const float*)d_in[10];
  const float* ln2_scale = (const float*)d_in[11];
  const float* ln2_bias  = (const float*)d_in[12];
  const float* W1        = (const float*)d_in[13];
  const float* b1        = (const float*)d_in[14];
  const float* W2        = (const float*)d_in[15];
  const float* b2        = (const float*)d_in[16];
  const float* lnf_scale = (const float*)d_in[17];
  const float* lnf_bias  = (const float*)d_in[18];
  const float* Wg        = (const float*)d_in[19];
  const float* bg        = (const float*)d_in[20];

  float* out = (float*)d_out;  // [N_NODES, VOCAB]

  // --- workspace carve-up ---
  char* w = (char*)d_ws;
  auto alloc = [&](size_t bytes) -> char* {
    char* p = w;
    w += (bytes + 255) & ~(size_t)255;
    return p;
  };
  float*    x      = (float*)   alloc((size_t)N_NODES * D_MODEL * 4);
  _Float16* xn_h   = (_Float16*)alloc((size_t)N_NODES * D_MODEL * 2);
  float*    qkv    = (float*)   alloc((size_t)N_NODES * 3 * D_MODEL * 4);
  _Float16* z_h    = (_Float16*)alloc((size_t)N_NODES * D_MODEL * 2);
  _Float16* ff1_h  = (_Float16*)alloc((size_t)N_NODES * D_FF * 2);
  _Float16* Wqkv_t = (_Float16*)alloc((size_t)N_LAYERS * 3 * D_MODEL * D_MODEL * 2);
  _Float16* Wo_t   = (_Float16*)alloc((size_t)N_LAYERS * D_MODEL * D_MODEL * 2);
  _Float16* W1_t   = (_Float16*)alloc((size_t)N_LAYERS * D_FF * D_MODEL * 2);
  _Float16* W2_t   = (_Float16*)alloc((size_t)N_LAYERS * D_MODEL * D_FF * 2);
  _Float16* Wg_t   = (_Float16*)alloc((size_t)VOCAB * D_MODEL * 2);

  auto cvt = [&](const float* in, _Float16* o, int K, int N) {
    int total = K * N;
    cvt_transpose_kernel<<<(total + 255) / 256, 256, 0, stream>>>(in, o, K, N);
  };

  // --- weight convert+transpose (deterministic, every call) ---
  for (int l = 0; l < N_LAYERS; ++l) {
    cvt(Wqkv + (size_t)l * D_MODEL * 3 * D_MODEL, Wqkv_t + (size_t)l * 3 * D_MODEL * D_MODEL,
        D_MODEL, 3 * D_MODEL);
    cvt(Wo   + (size_t)l * D_MODEL * D_MODEL,     Wo_t   + (size_t)l * D_MODEL * D_MODEL,
        D_MODEL, D_MODEL);
    cvt(W1   + (size_t)l * D_MODEL * D_FF,        W1_t   + (size_t)l * D_FF * D_MODEL,
        D_MODEL, D_FF);
    cvt(W2   + (size_t)l * D_FF * D_MODEL,        W2_t   + (size_t)l * D_MODEL * D_FF,
        D_FF, D_MODEL);
  }
  cvt(Wg, Wg_t, D_MODEL, VOCAB);

  // --- embedding ---
  embed_kernel<<<(N_NODES * D_MODEL) / 256, 256, 0, stream>>>(
      tokens, positions, coord_emb, pos_emb, value_emb, x);

  // --- layers ---
  for (int l = 0; l < N_LAYERS; ++l) {
    const float* l1s = ln1_scale + l * D_MODEL;
    const float* l1b = ln1_bias  + l * D_MODEL;
    const float* l2s = ln2_scale + l * D_MODEL;
    const float* l2b = ln2_bias  + l * D_MODEL;

    // xn = LN1(x) -> f16
    ln_kernel<<<N_NODES / 8, 256, 0, stream>>>(x, l1s, l1b, xn_h);
    // qkv = xn @ Wqkv  (f32)
    launch_gemm(xn_h, Wqkv_t + (size_t)l * 3 * D_MODEL * D_MODEL,
                nullptr, nullptr, qkv, nullptr,
                N_NODES, 3 * D_MODEL, D_MODEL, 0, stream);
    // z = causal attention(qkv) -> f16
    attn_kernel<<<N_NODES, NHEAD * 32, 0, stream>>>(qkv, z_h);
    // x = x + z @ Wo
    launch_gemm(z_h, Wo_t + (size_t)l * D_MODEL * D_MODEL,
                nullptr, x, x, nullptr,
                N_NODES, D_MODEL, D_MODEL, 0, stream);
    // h = LN2(x) -> f16
    ln_kernel<<<N_NODES / 8, 256, 0, stream>>>(x, l2s, l2b, xn_h);
    // ff1 = relu(h @ W1 + b1) -> f16
    launch_gemm(xn_h, W1_t + (size_t)l * D_FF * D_MODEL,
                b1 + (size_t)l * D_FF, nullptr, nullptr, ff1_h,
                N_NODES, D_FF, D_MODEL, 1, stream);
    // x = x + ff1 @ W2 + b2
    launch_gemm(ff1_h, W2_t + (size_t)l * D_MODEL * D_FF,
                b2 + (size_t)l * D_MODEL, x, x, nullptr,
                N_NODES, D_MODEL, D_FF, 0, stream);
  }

  // --- head ---
  ln_kernel<<<N_NODES / 8, 256, 0, stream>>>(x, lnf_scale, lnf_bias, xn_h);
  launch_gemm(xn_h, Wg_t, bg, nullptr, out, nullptr,
              N_NODES, VOCAB, D_MODEL, 0, stream);
  logsoftmax_kernel<<<N_NODES / 8, 256, 0, stream>>>(out, VOCAB);
}
